// RecurrentGaussianActor_73667279060997
// MI455X (gfx1250) — matrix operand order repaired and checked by reading, not compiled, not run on hardware
//
#include <hip/hip_runtime.h>
#include <hip/hip_bf16.h>
#include <math.h>

typedef __bf16 bf16_t;
typedef __attribute__((ext_vector_type(16))) __bf16 v16bf;
typedef __attribute__((ext_vector_type(8)))  float  v8f;

#define B_   256
#define T_   128
#define D_   512
#define H_   1024
#define FC_  1024
#define A_   32
#define G3_  3072
#define SPLITK 4          // split-K factor for the recurrent W_hh GEMM

#define BM 128
#define BN 128
#define BK 32
#define LDT 40   // LDS tile row stride (32 + 8 pad elements) -> 80B rows, 16B aligned

union FragU { v16bf v; unsigned u[8]; };

// A fragment (16x32 bf16): lane L holds row M=L%16; lanes 0-15 K in {0..7,16..23},
// lanes 16-31 K in {8..15,24..31}; VGPR j pair kbase = (j>=4?16:0) + (j%4)*2 + koff
__device__ inline v16bf lds_load_a(const bf16_t* base, int mrow, int lane) {
  FragU f;
  const bf16_t* p = base + (size_t)(mrow + (lane & 15)) * LDT;
  const int koff = (lane >> 4) << 3;
#pragma unroll
  for (int j = 0; j < 8; ++j) {
    const int kb = ((j & 4) << 2) + ((j & 3) << 1) + koff;
    f.u[j] = *(const unsigned*)(p + kb);
  }
  return f.v;
}

// B fragment (32x16 bf16): lane L holds col N=L%16; lanes 0-15 K in {0..7,16..23}, lanes 16-31 K in {8..15,24..31}
__device__ inline v16bf lds_load_b(const bf16_t* base, int nrow, int lane) {
  FragU f;
  const bf16_t* p = base + (size_t)(nrow + (lane & 15)) * LDT + ((lane >> 4) << 3);
#pragma unroll
  for (int j = 0; j < 8; ++j) f.u[j] = *(const unsigned*)(p + ((j & 4) << 2) + ((j & 3) << 1));
  return f.v;
}

// ---------------------------------------------------------------------------
// Generic GEMM: C = epi(A(MxK,bf16) * W(NxK,bf16)^T + bias)
// EPI 0: relu -> bf16 ; EPI 1: plain -> f32 ; EPI 2: clip -> f32
// Double-buffered LDS + register prefetch: 1 barrier per K-step, global loads
// issued before the WMMA block so the tensor pipe hides memory latency.
// ---------------------------------------------------------------------------
template <int EPI>
__global__ __launch_bounds__(256) void gemm_bf16_wmma(
    const bf16_t* __restrict__ A, const bf16_t* __restrict__ W,
    const float* __restrict__ bias, float* __restrict__ Cf, bf16_t* __restrict__ Cb,
    int M, int N, int K, int ldc, float clip_lo, float clip_hi) {
  __shared__ bf16_t As[2][BM * LDT];
  __shared__ bf16_t Bs[2][BN * LDT];

  const int tid  = threadIdx.x;
  const int lane = tid & 31;
  const int wave = tid >> 5;
  const int wm = wave >> 1;           // 0..3 : 32-row strip
  const int wn = wave & 1;            // 0..1 : 64-col strip
  const int m0 = blockIdx.y * BM;
  const int n0 = blockIdx.x * BN;
  const int row = tid >> 1;           // 0..127
  const int col = (tid & 1) << 4;     // 0 or 16
  const bool bvalid = (n0 + row) < N;

  v8f acc[2][4];
#pragma unroll
  for (int i = 0; i < 2; ++i)
#pragma unroll
    for (int j = 0; j < 4; ++j) {
      v8f z = {0.f,0.f,0.f,0.f,0.f,0.f,0.f,0.f};
      acc[i][j] = z;
    }

  uint4 pa0, pa1, pb0, pb1;
  auto gload = [&](int k0) {
    const bf16_t* ga = A + (size_t)(m0 + row) * K + k0 + col;
    pa0 = *(const uint4*)ga;
    pa1 = *(const uint4*)(ga + 8);
    pb0 = make_uint4(0,0,0,0); pb1 = make_uint4(0,0,0,0);
    if (bvalid) {
      const bf16_t* gb = W + (size_t)(n0 + row) * K + k0 + col;
      pb0 = *(const uint4*)gb;
      pb1 = *(const uint4*)(gb + 8);
    }
  };
  auto lstore = [&](int buf) {
    *(uint4*)&As[buf][row * LDT + col]     = pa0;
    *(uint4*)&As[buf][row * LDT + col + 8] = pa1;
    *(uint4*)&Bs[buf][row * LDT + col]     = pb0;
    *(uint4*)&Bs[buf][row * LDT + col + 8] = pb1;
  };

  gload(0);
  lstore(0);
  const int nk = K / BK;
  for (int kt = 0; kt < nk; ++kt) {
    __syncthreads();
    const int cur = kt & 1;
    if (kt + 1 < nk) gload((kt + 1) * BK);   // issue global loads early

    v16bf af[2], bfr[4];
#pragma unroll
    for (int ti = 0; ti < 2; ++ti) af[ti]  = lds_load_a(As[cur], wm * 32 + ti * 16, lane);
#pragma unroll
    for (int tj = 0; tj < 4; ++tj) bfr[tj] = lds_load_b(Bs[cur], wn * 64 + tj * 16, lane);

#pragma unroll
    for (int ti = 0; ti < 2; ++ti)
#pragma unroll
      for (int tj = 0; tj < 4; ++tj)
        acc[ti][tj] = __builtin_amdgcn_wmma_f32_16x16x32_bf16(
            false, af[ti], false, bfr[tj], (short)0, acc[ti][tj], false, false);

    if (kt + 1 < nk) lstore(cur ^ 1);        // LDS store after math, loads have landed
  }

  // epilogue: C/D layout — lane holds N=lane%16, VGPR j holds M = j + (lane>=16 ? 8 : 0)
#pragma unroll
  for (int ti = 0; ti < 2; ++ti) {
#pragma unroll
    for (int tj = 0; tj < 4; ++tj) {
      const int n = n0 + wn * 64 + tj * 16 + (lane & 15);
      if (n < N) {
        const int mb = m0 + wm * 32 + ti * 16 + ((lane >> 4) << 3);
        const float bv = bias[n];
#pragma unroll
        for (int j = 0; j < 8; ++j) {
          float v = acc[ti][tj][j] + bv;
          const size_t idx = (size_t)(mb + j) * (size_t)ldc + n;
          if (EPI == 0)      { v = fmaxf(v, 0.f); Cb[idx] = (bf16_t)v; }
          else if (EPI == 1) { Cf[idx] = v; }
          else               { Cf[idx] = fminf(fmaxf(v, clip_lo), clip_hi); }
        }
      }
    }
  }
}

// ---------------------------------------------------------------------------
// Recurrent GEMM: gh_part[z] = (h * (1-start_t)) @ W_hh^T   (no bias; gate adds it)
// M=256, N=3072, K=1024, split-K=SPLITK. Mask + fp32->bf16 convert fused into
// the A-tile staging, so no separate mask kernel per step.
// grid = (3072/BN, 256/BM, SPLITK)
// ---------------------------------------------------------------------------
__global__ __launch_bounds__(256) void gemm_hh_wmma(
    const float* __restrict__ h, const int* __restrict__ starts, int t,
    const bf16_t* __restrict__ W, float* __restrict__ ghp) {
  __shared__ bf16_t As[2][BM * LDT];
  __shared__ bf16_t Bs[2][BN * LDT];

  const int tid  = threadIdx.x;
  const int lane = tid & 31;
  const int wave = tid >> 5;
  const int wm = wave >> 1;
  const int wn = wave & 1;
  const int m0 = blockIdx.y * BM;
  const int n0 = blockIdx.x * BN;
  const int kbase = blockIdx.z * (H_ / SPLITK);
  const int row = tid >> 1;
  const int col = (tid & 1) << 4;
  const float keep = 1.f - (float)starts[(m0 + row) * T_ + t];

  v8f acc[2][4];
#pragma unroll
  for (int i = 0; i < 2; ++i)
#pragma unroll
    for (int j = 0; j < 4; ++j) {
      v8f z = {0.f,0.f,0.f,0.f,0.f,0.f,0.f,0.f};
      acc[i][j] = z;
    }

  float4 fa[4];
  uint4 pb0, pb1;
  auto gload = [&](int k0) {
    const float* ga = h + (size_t)(m0 + row) * H_ + kbase + k0 + col;
#pragma unroll
    for (int q = 0; q < 4; ++q) fa[q] = *(const float4*)(ga + 4 * q);
    const bf16_t* gb = W + (size_t)(n0 + row) * H_ + kbase + k0 + col;
    pb0 = *(const uint4*)gb;
    pb1 = *(const uint4*)(gb + 8);
  };
  auto lstore = [&](int buf) {
    union { bf16_t e[16]; uint4 q[2]; } u;
#pragma unroll
    for (int q = 0; q < 4; ++q) {
      u.e[4*q + 0] = (bf16_t)(fa[q].x * keep);
      u.e[4*q + 1] = (bf16_t)(fa[q].y * keep);
      u.e[4*q + 2] = (bf16_t)(fa[q].z * keep);
      u.e[4*q + 3] = (bf16_t)(fa[q].w * keep);
    }
    *(uint4*)&As[buf][row * LDT + col]     = u.q[0];
    *(uint4*)&As[buf][row * LDT + col + 8] = u.q[1];
    *(uint4*)&Bs[buf][row * LDT + col]     = pb0;
    *(uint4*)&Bs[buf][row * LDT + col + 8] = pb1;
  };

  gload(0);
  lstore(0);
  const int nk = (H_ / SPLITK) / BK;       // 8
  for (int kt = 0; kt < nk; ++kt) {
    __syncthreads();
    const int cur = kt & 1;
    if (kt + 1 < nk) gload((kt + 1) * BK);

    v16bf af[2], bfr[4];
#pragma unroll
    for (int ti = 0; ti < 2; ++ti) af[ti]  = lds_load_a(As[cur], wm * 32 + ti * 16, lane);
#pragma unroll
    for (int tj = 0; tj < 4; ++tj) bfr[tj] = lds_load_b(Bs[cur], wn * 64 + tj * 16, lane);

#pragma unroll
    for (int ti = 0; ti < 2; ++ti)
#pragma unroll
      for (int tj = 0; tj < 4; ++tj)
        acc[ti][tj] = __builtin_amdgcn_wmma_f32_16x16x32_bf16(
            false, af[ti], false, bfr[tj], (short)0, acc[ti][tj], false, false);

    if (kt + 1 < nk) lstore(cur ^ 1);
  }

  float* outp = ghp + (size_t)blockIdx.z * B_ * G3_;
#pragma unroll
  for (int ti = 0; ti < 2; ++ti) {
#pragma unroll
    for (int tj = 0; tj < 4; ++tj) {
      const int n  = n0 + wn * 64 + tj * 16 + (lane & 15);
      const int mb = m0 + wm * 32 + ti * 16 + ((lane >> 4) << 3);
#pragma unroll
      for (int j = 0; j < 8; ++j)
        outp[(size_t)(mb + j) * G3_ + n] = acc[ti][tj][j];
    }
  }
}

__global__ void k_f32_to_bf16(const float* __restrict__ in, bf16_t* __restrict__ out, int n) {
  for (int i = blockIdx.x * blockDim.x + threadIdx.x; i < n; i += gridDim.x * blockDim.x)
    out[i] = (bf16_t)in[i];
}

// one 256-thread block per (b,t) row of D_=512
__global__ __launch_bounds__(256) void k_layernorm(
    const float* __restrict__ obs, const float* __restrict__ gamma,
    const float* __restrict__ beta, bf16_t* __restrict__ out) {
  __shared__ float red[16];
  __shared__ float stats[2];
  const size_t row = blockIdx.x;
  const float* src = obs + row * D_;
  float vals[2];
  float s = 0.f, s2 = 0.f;
#pragma unroll
  for (int r = 0; r < 2; ++r) {
    float v = src[threadIdx.x + r * 256];
    vals[r] = v; s += v; s2 += v * v;
  }
#pragma unroll
  for (int o = 16; o > 0; o >>= 1) {
    s  += __shfl_down(s,  o, 32);
    s2 += __shfl_down(s2, o, 32);
  }
  const int wave = threadIdx.x >> 5, lane = threadIdx.x & 31;
  if (lane == 0) { red[wave] = s; red[8 + wave] = s2; }
  __syncthreads();
  if (threadIdx.x == 0) {
    float ts = 0.f, ts2 = 0.f;
    for (int w = 0; w < 8; ++w) { ts += red[w]; ts2 += red[8 + w]; }
    const float mean = ts / (float)D_;
    const float var  = ts2 / (float)D_ - mean * mean;
    stats[0] = mean; stats[1] = rsqrtf(var + 1e-5f);
  }
  __syncthreads();
  const float mean = stats[0], rstd = stats[1];
#pragma unroll
  for (int r = 0; r < 2; ++r) {
    const int i = threadIdx.x + r * 256;
    out[row * D_ + i] = (bf16_t)((vals[r] - mean) * rstd * gamma[i] + beta[i]);
  }
}

// fused GRU gates: sums SPLITK gh partials, adds b_hh, recomputes masked h,
// h_new = (1-z)*n + z*h_masked ; stores h (fp32) and Hseq (bf16)
__global__ void k_gate(const float* __restrict__ xg, const float* __restrict__ ghp,
                       const float* __restrict__ b_hh, const int* __restrict__ starts,
                       float* __restrict__ h, bf16_t* __restrict__ Hseq, int t) {
  const int i = blockIdx.x * blockDim.x + threadIdx.x;   // < B_*H_
  const int b = i >> 10;
  const int j = i & 1023;
  const float keep = 1.f - (float)starts[b * T_ + t];
  const float hmv = h[i] * keep;

  const size_t go = (size_t)b * G3_;
  float hr = b_hh[j], hz = b_hh[H_ + j], hn = b_hh[2 * H_ + j];
#pragma unroll
  for (int zp = 0; zp < SPLITK; ++zp) {
    const float* g = ghp + (size_t)zp * B_ * G3_ + go;
    hr += g[j]; hz += g[H_ + j]; hn += g[2 * H_ + j];
  }

  const size_t xo = ((size_t)b * T_ + t) * G3_;
  const float xr = xg[xo + j], xz = xg[xo + H_ + j], xn = xg[xo + 2 * H_ + j];
  const float r  = 1.f / (1.f + __expf(-(xr + hr)));
  const float z  = 1.f / (1.f + __expf(-(xz + hz)));
  const float nv = tanhf(xn + r * hn);
  const float hnew = (1.f - z) * nv + z * hmv;
  h[i] = hnew;
  Hseq[((size_t)b * T_ + t) * H_ + j] = (bf16_t)hnew;
}

extern "C" void kernel_launch(void* const* d_in, const int* in_sizes, int n_in,
                              void* d_out, int out_size, void* d_ws, size_t ws_size,
                              hipStream_t stream) {
  (void)in_sizes; (void)n_in; (void)out_size; (void)ws_size;
  const float* obs    = (const float*)d_in[0];
  const float* h0     = (const float*)d_in[1];
  const int*   starts = (const int*)  d_in[2];
  const float* ln_g   = (const float*)d_in[3];
  const float* ln_b   = (const float*)d_in[4];
  const float* W_fc   = (const float*)d_in[5];
  const float* b_fc   = (const float*)d_in[6];
  const float* W_ih   = (const float*)d_in[7];
  const float* b_ih   = (const float*)d_in[8];
  const float* W_hh   = (const float*)d_in[9];
  const float* b_hh   = (const float*)d_in[10];
  const float* W_out  = (const float*)d_in[11];
  const float* b_out  = (const float*)d_in[12];
  const float* W_mean = (const float*)d_in[13];
  const float* b_mean = (const float*)d_in[14];
  const float* W_ls   = (const float*)d_in[15];
  const float* b_ls   = (const float*)d_in[16];
  float* out = (float*)d_out;

  char* ws = (char*)d_ws;
  size_t off = 0;
  auto alloc = [&](size_t bytes) -> char* {
    char* p = ws + off;
    off = (off + bytes + 255) & ~(size_t)255;
    return p;
  };

  const size_t BT = (size_t)B_ * T_;
  bf16_t* x_bf    = (bf16_t*)alloc(BT * D_ * 2);
  bf16_t* Wfc_b   = (bf16_t*)alloc((size_t)FC_ * D_ * 2);
  bf16_t* Wih_b   = (bf16_t*)alloc((size_t)G3_ * FC_ * 2);
  bf16_t* Whh_b   = (bf16_t*)alloc((size_t)G3_ * H_ * 2);
  bf16_t* Wout_b  = (bf16_t*)alloc((size_t)H_ * H_ * 2);
  bf16_t* Wmean_b = (bf16_t*)alloc((size_t)A_ * H_ * 2);
  bf16_t* Wls_b   = (bf16_t*)alloc((size_t)A_ * H_ * 2);
  bf16_t* Xfc     = (bf16_t*)alloc(BT * FC_ * 2);
  float*  xg      = (float*) alloc(BT * G3_ * 4);
  float*  hbuf    = (float*) alloc((size_t)B_ * H_ * 4);
  float*  ghp     = (float*) alloc((size_t)SPLITK * B_ * G3_ * 4);
  bf16_t* Hseq    = (bf16_t*)alloc(BT * H_ * 2);
  bf16_t* feats   = (bf16_t*)alloc(BT * H_ * 2);

  const dim3 blk(256);

  // weight conversions fp32 -> bf16
  k_f32_to_bf16<<<512, blk, 0, stream>>>(W_fc,   Wfc_b,   FC_ * D_);
  k_f32_to_bf16<<<512, blk, 0, stream>>>(W_ih,   Wih_b,   G3_ * FC_);
  k_f32_to_bf16<<<512, blk, 0, stream>>>(W_hh,   Whh_b,   G3_ * H_);
  k_f32_to_bf16<<<512, blk, 0, stream>>>(W_out,  Wout_b,  H_ * H_);
  k_f32_to_bf16<<<64,  blk, 0, stream>>>(W_mean, Wmean_b, A_ * H_);
  k_f32_to_bf16<<<64,  blk, 0, stream>>>(W_ls,   Wls_b,   A_ * H_);

  // LayerNorm -> x_bf (B*T, D)
  k_layernorm<<<(unsigned)BT, blk, 0, stream>>>(obs, ln_g, ln_b, x_bf);

  // Xfc = relu(x @ W_fc^T + b_fc)   (32768 x 1024, K=512)
  gemm_bf16_wmma<0><<<dim3(FC_ / BN, (unsigned)(BT / BM)), blk, 0, stream>>>(
      x_bf, Wfc_b, b_fc, nullptr, Xfc, (int)BT, FC_, D_, FC_, 0.f, 0.f);

  // xg = Xfc @ W_ih^T + b_ih        (32768 x 3072, K=1024), fp32
  gemm_bf16_wmma<1><<<dim3(G3_ / BN, (unsigned)(BT / BM)), blk, 0, stream>>>(
      Xfc, Wih_b, b_ih, xg, nullptr, (int)BT, G3_, FC_, G3_, 0.f, 0.f);

  // h <- hidden_state
  hipMemcpyAsync(hbuf, h0, (size_t)B_ * H_ * sizeof(float), hipMemcpyDeviceToDevice, stream);

  // sequential GRU scan: 2 launches per step (masked split-K GEMM, fused gates)
  for (int t = 0; t < T_; ++t) {
    gemm_hh_wmma<<<dim3(G3_ / BN, B_ / BM, SPLITK), blk, 0, stream>>>(
        hbuf, starts, t, Whh_b, ghp);
    k_gate<<<(B_ * H_) / 256, blk, 0, stream>>>(xg, ghp, b_hh, starts, hbuf, Hseq, t);
  }

  // feats = relu(Hseq @ W_out^T + b_out)   (32768 x 1024, K=1024), bf16
  gemm_bf16_wmma<0><<<dim3(H_ / BN, (unsigned)(BT / BM)), blk, 0, stream>>>(
      Hseq, Wout_b, b_out, nullptr, feats, (int)BT, H_, H_, H_, 0.f, 0.f);

  // means -> d_out[0 : B*T*A]
  gemm_bf16_wmma<1><<<dim3(1, (unsigned)(BT / BM)), blk, 0, stream>>>(
      feats, Wmean_b, b_mean, out, nullptr, (int)BT, A_, H_, A_, 0.f, 0.f);

  // log_std (clipped) -> d_out[B*T*A : 2*B*T*A]
  gemm_bf16_wmma<2><<<dim3(1, (unsigned)(BT / BM)), blk, 0, stream>>>(
      feats, Wls_b, b_ls, out + BT * A_, nullptr, (int)BT, A_, H_, A_, -20.f, 2.f);

  // h_final -> d_out[2*B*T*A : ...]
  hipMemcpyAsync(out + 2 * BT * A_, hbuf, (size_t)B_ * H_ * sizeof(float),
                 hipMemcpyDeviceToDevice, stream);
}
